// fakeFCGQCNN_22866405884157
// MI455X (gfx1250) — compile-verified, matrix operand order checked by reading, not run on hardware
//
#include <hip/hip_runtime.h>
#include <hip/hip_bf16.h>
#include <math.h>

// ---------------------------------------------------------------------------
// Fully-convolutional GQCNN for MI455X (gfx1250, wave32, WMMA).
//
// Shared-trunk rewrite:
//   conv1 (5x5,16) full image        -> [2,16,92,92]
//   pool1 stride-1 2x2               -> [2,16,91,91]
//   conv2 3x3 dilation-2             -> [2,32,87,87]
//   pool2 stride-1 dilation-2 2x2    -> [2,32,85,85]  (stored as f16)
//   fc1 as one GEMM [8450 x 1152] x [1152 x 128] via v_wmma_f32_16x16x32_f16
//     with 4-way M register blocking (each B fragment feeds 4 WMMAs,
//     cutting L2 weight traffic 4x: ~156 MB -> ~39 MB).
//   fused fc2 + sigmoid epilogue, scatter into 96x96 heatmap interior.
//
// This revision hoists the four per-M-tile LDS base pointers out of the
// k-loop so every ds_load_b128 uses a 16-bit immediate offset (tile strides
// 2 and 3 exceeded the DS immediate range before), removing per-iteration
// v_add address math and the WMMA->VALU hazard v_nops it forced.
// ---------------------------------------------------------------------------

typedef __attribute__((ext_vector_type(16))) _Float16 v16h;
typedef __attribute__((ext_vector_type(8)))  float    v8f;

// Problem constants (match setup_inputs()).
constexpr int B    = 2;
constexpr int HW   = 96;
constexpr int H1   = 92;                 // 96-5+1
constexpr int P1   = 91;                 // stride-1 pool of 92
constexpr int H2   = 87;                 // 91 - (2*2+1) + 1  (3x3 dil-2)
constexpr int P2   = 85;                 // stride-1 dil-2 pool of 87
constexpr int HO   = 65;                 // crops per dim
constexpr int NTOT = B * HO * HO;        // 8450 crops
constexpr int KDIM = 32 * 6 * 6;         // 1152 fc1 inputs
constexpr int NOUT = 128;                // fc1 outputs
constexpr int KSTEPS = KDIM / 32;        // 36 wmma k-steps
constexpr int MBLK  = 4;                 // M-tiles per block (register blocking)
constexpr int MROWS = 16 * MBLK;         // 64 crops per block
constexpr int NBLK_GEMM = (NTOT + MROWS - 1) / MROWS;  // 133
constexpr int LDSROW = KDIM + 16;        // 1168 halfs -> 2336 B row stride (32B multiple)

constexpr int N_CONV1 = B * 16 * H1 * H1;
constexpr int N_POOL1 = B * 16 * P1 * P1;
constexpr int N_CONV2 = B * 32 * H2 * H2;
constexpr int N_POOL2 = B * 32 * P2 * P2;
constexpr int N_WCVT  = NOUT * KDIM;

// ---------------- trunk kernels (cheap, VALU) ------------------------------

__global__ void conv1_relu_k(const float* __restrict__ x,
                             const float* __restrict__ w1,
                             const float* __restrict__ b1,
                             float* __restrict__ out) {
  int idx = blockIdx.x * blockDim.x + threadIdx.x;
  if (idx >= N_CONV1) return;
  int xx = idx % H1;
  int t  = idx / H1;
  int yy = t % H1;  t /= H1;
  int o  = t % 16;
  int b  = t / 16;
  const float* xi = x + b * HW * HW;
  const float* wv = w1 + o * 25;
  float acc = b1[o];
#pragma unroll
  for (int dy = 0; dy < 5; ++dy)
#pragma unroll
    for (int dx = 0; dx < 5; ++dx)
      acc = fmaf(xi[(yy + dy) * HW + (xx + dx)], wv[dy * 5 + dx], acc);
  out[idx] = fmaxf(acc, 0.0f);
}

__global__ void pool1_k(const float* __restrict__ c1, float* __restrict__ out) {
  int idx = blockIdx.x * blockDim.x + threadIdx.x;
  if (idx >= N_POOL1) return;
  int x = idx % P1;
  int t = idx / P1;
  int y = t % P1;  t /= P1;
  int c = t % 16;
  int b = t / 16;
  const float* s = c1 + ((b * 16 + c) * H1 + y) * H1 + x;
  float m = fmaxf(fmaxf(s[0], s[1]), fmaxf(s[H1], s[H1 + 1]));
  out[idx] = m;
}

__global__ void conv2_dil2_relu_k(const float* __restrict__ m,
                                  const float* __restrict__ w2,
                                  const float* __restrict__ b2,
                                  float* __restrict__ out) {
  int idx = blockIdx.x * blockDim.x + threadIdx.x;
  if (idx >= N_CONV2) return;
  int x = idx % H2;
  int t = idx / H2;
  int y = t % H2;  t /= H2;
  int o = t % 32;
  int b = t / 32;
  float acc = b2[o];
  for (int c = 0; c < 16; ++c) {
    const float* mp = m + ((b * 16 + c) * P1 + y) * P1 + x;
    const float* wp = w2 + (o * 16 + c) * 9;
#pragma unroll
    for (int dy = 0; dy < 3; ++dy)
#pragma unroll
      for (int dx = 0; dx < 3; ++dx)
        acc = fmaf(mp[dy * 2 * P1 + dx * 2], wp[dy * 3 + dx], acc);
  }
  out[idx] = fmaxf(acc, 0.0f);
}

// Pooled feature map is the WMMA A-matrix source: emit f16 directly so the
// GEMM staging gather is half the bytes and needs no convert.
__global__ void pool2_dil2_f16_k(const float* __restrict__ c2,
                                 _Float16* __restrict__ out) {
  int idx = blockIdx.x * blockDim.x + threadIdx.x;
  if (idx >= N_POOL2) return;
  int x = idx % P2;
  int t = idx / P2;
  int y = t % P2;  t /= P2;
  int c = t % 32;
  int b = t / 32;
  const float* s = c2 + ((b * 32 + c) * H2 + y) * H2 + x;
  float m = fmaxf(fmaxf(s[0], s[2]), fmaxf(s[2 * H2], s[2 * H2 + 2]));
  out[idx] = (_Float16)m;
}

__global__ void cvt_w_f16_k(const float* __restrict__ w, _Float16* __restrict__ wf16) {
  int idx = blockIdx.x * blockDim.x + threadIdx.x;
  if (idx < N_WCVT) wf16[idx] = (_Float16)w[idx];
}

__global__ void zero_out_k(float* __restrict__ out, int n) {
  int idx = blockIdx.x * blockDim.x + threadIdx.x;
  if (idx < n) out[idx] = 0.0f;
}

// ---------------- fused fc1 (WMMA) + fc2 + sigmoid -------------------------
//
// One block per 64-crop super-tile (4 M-tiles). 8 wave32s; wave w owns fc1
// output columns [16w, 16w+16) for ALL 4 M-tiles (4 accumulators), so each
// B fragment (global) is reused by 4 v_wmma ops.
//
// A-fragment (16-bit A 16x32, this code's contiguous k order, matching B): lane L -> M = L%16, K base = 16*(L/16),
//   contiguous 16 halfs -> one aligned v16h LDS read.
// B-fragment (32x16, K x N): lane L -> N = L%16, K base = 16*(L/16),
//   contiguous 16 halfs of fc1_w[o][k] -> one aligned v16h global read.
// D/C (16x16 f32): lane L -> col N = L%16; VGPR r -> row M = r + 8*(L/16).

__global__ __launch_bounds__(256) void gemm_wmma_head_k(
    const _Float16* __restrict__ p2,
    const _Float16* __restrict__ wf16,
    const float* __restrict__ fc1_b,
    const float* __restrict__ fc2_w,
    const float* __restrict__ fc2_b,
    float* __restrict__ out) {
  __shared__ __align__(32) _Float16 A[MROWS][LDSROW];   // ~146 KB (CDNA5: 320 KB/WGP)
  __shared__ float acc[MROWS];

  const int tid = threadIdx.x;
  const int n0  = blockIdx.x * MROWS;    // first crop row of this block

  for (int i = tid; i < MROWS; i += 256) acc[i] = 0.0f;

  // Stage the 64 x 1152 feature tile (strided f16 gather from pooled map).
  for (int idx = tid; idx < MROWS * KDIM; idx += 256) {
    int ml = idx / KDIM;                 // 0..63 (tile row)
    int k  = idx - ml * KDIM;
    int n  = n0 + ml;
    if (n >= NTOT) n = 0;               // padded rows: value irrelevant, never written
    int j  = n % HO;
    int t2 = n / HO;
    int i2 = t2 % HO;
    int b  = t2 / HO;
    int c  = k / 36;
    int r  = k - c * 36;
    int tt = r / 6;
    int ss = r - tt * 6;
    A[ml][k] = p2[((b * 32 + c) * P2 + (i2 + 4 * tt)) * P2 + (j + 4 * ss)];
  }
  __syncthreads();

  const int lane = tid & 31;
  const int wv   = tid >> 5;      // wave id == N tile
  const int lo   = lane & 15;
  const int hi   = lane >> 4;
  const int o    = wv * 16 + lo;  // fc1 output channel owned by this lane

  const _Float16* __restrict__ Bbase = wf16 + o * KDIM + hi * 16;
  // Hoisted, loop-invariant per-tile LDS bases: every ds_load in the k-loop
  // becomes base + imm (imm <= 2288 B, well inside the 16-bit DS offset).
  const _Float16* __restrict__ A0 = &A[0 * 16 + lo][hi * 16];
  const _Float16* __restrict__ A1 = &A[1 * 16 + lo][hi * 16];
  const _Float16* __restrict__ A2 = &A[2 * 16 + lo][hi * 16];
  const _Float16* __restrict__ A3 = &A[3 * 16 + lo][hi * 16];
  __builtin_prefetch(Bbase, 0, 1);

  v8f C0 = {}, C1 = {}, C2 = {}, C3 = {};
#pragma unroll
  for (int kk = 0; kk < KSTEPS; ++kk) {
    v16h b  = *(const v16h*)(Bbase + kk * 32);
    v16h a0 = *(const v16h*)(A0 + kk * 32);
    v16h a1 = *(const v16h*)(A1 + kk * 32);
    v16h a2 = *(const v16h*)(A2 + kk * 32);
    v16h a3 = *(const v16h*)(A3 + kk * 32);
    C0 = __builtin_amdgcn_wmma_f32_16x16x32_f16(false, a0, false, b, (short)0, C0, false, false);
    C1 = __builtin_amdgcn_wmma_f32_16x16x32_f16(false, a1, false, b, (short)0, C1, false, false);
    C2 = __builtin_amdgcn_wmma_f32_16x16x32_f16(false, a2, false, b, (short)0, C2, false, false);
    C3 = __builtin_amdgcn_wmma_f32_16x16x32_f16(false, a3, false, b, (short)0, C3, false, false);
  }

  // Epilogue: h = relu(C + fc1_b[o]); partial fc2 dot = h * fc2_w[o];
  // reduce across the 16 lanes of each half, then across waves via LDS atomics.
  const float bias = fc1_b[o];
  const float w2o  = fc2_w[o];
  v8f Cm[MBLK] = {C0, C1, C2, C3};
#pragma unroll
  for (int mt = 0; mt < MBLK; ++mt) {
#pragma unroll
    for (int r = 0; r < 8; ++r) {
      float p = fmaxf(Cm[mt][r] + bias, 0.0f) * w2o;
      p += __shfl_xor(p, 1);
      p += __shfl_xor(p, 2);
      p += __shfl_xor(p, 4);
      p += __shfl_xor(p, 8);
      if (lo == 0) atomicAdd(&acc[mt * 16 + hi * 8 + r], p);  // row = 8*hi + r
    }
  }
  __syncthreads();

  if (tid < MROWS) {
    int n = n0 + tid;
    if (n < NTOT) {
      int j  = n % HO;
      int t2 = n / HO;
      int i2 = t2 % HO;
      int b  = t2 / HO;
      float z = acc[tid] + fc2_b[0];
      float q = 1.0f / (1.0f + expf(-z));
      out[(b * HW + (16 + i2)) * HW + (16 + j)] = q;
    }
  }
}

// ---------------------------------------------------------------------------

extern "C" void kernel_launch(void* const* d_in, const int* in_sizes, int n_in,
                              void* d_out, int out_size, void* d_ws, size_t ws_size,
                              hipStream_t stream) {
  const float* x     = (const float*)d_in[0];
  const float* w1    = (const float*)d_in[1];
  const float* b1    = (const float*)d_in[2];
  const float* w2    = (const float*)d_in[3];
  const float* b2    = (const float*)d_in[4];
  const float* fc1_w = (const float*)d_in[5];
  const float* fc1_b = (const float*)d_in[6];
  const float* fc2_w = (const float*)d_in[7];
  const float* fc2_b = (const float*)d_in[8];
  float* out = (float*)d_out;

  // Workspace carve-out (256B aligned slabs), ~5.4 MB total.
  char* ws = (char*)d_ws;
  size_t off = 0;
  auto take = [&](size_t bytes) -> char* {
    char* p = ws + off;
    off = (off + bytes + 255) & ~(size_t)255;
    return p;
  };
  float*    d_c1   = (float*)take((size_t)N_CONV1 * sizeof(float));
  float*    d_m    = (float*)take((size_t)N_POOL1 * sizeof(float));
  float*    d_c2   = (float*)take((size_t)N_CONV2 * sizeof(float));
  _Float16* d_p2   = (_Float16*)take((size_t)N_POOL2 * sizeof(_Float16));
  _Float16* d_wf16 = (_Float16*)take((size_t)N_WCVT * sizeof(_Float16));
  (void)ws_size; (void)in_sizes; (void)n_in;

  const int T = 256;
  zero_out_k<<<(out_size + T - 1) / T, T, 0, stream>>>(out, out_size);
  conv1_relu_k<<<(N_CONV1 + T - 1) / T, T, 0, stream>>>(x, w1, b1, d_c1);
  pool1_k<<<(N_POOL1 + T - 1) / T, T, 0, stream>>>(d_c1, d_m);
  conv2_dil2_relu_k<<<(N_CONV2 + T - 1) / T, T, 0, stream>>>(d_m, w2, b2, d_c2);
  pool2_dil2_f16_k<<<(N_POOL2 + T - 1) / T, T, 0, stream>>>(d_c2, d_p2);
  cvt_w_f16_k<<<(N_WCVT + T - 1) / T, T, 0, stream>>>(fc1_w, d_wf16);
  gemm_wmma_head_k<<<NBLK_GEMM, T, 0, stream>>>(d_p2, d_wf16, fc1_b, fc2_w, fc2_b, out);
}
